// MeshAEEncoder_53566832115904
// MI455X (gfx1250) — matrix operand order repaired and checked by reading, not compiled, not run on hardware
//
#include <hip/hip_runtime.h>

// ---------------------------------------------------------------------------
// MeshAE encoder for MI455X (gfx1250, wave32).
// All GEMMs run through one tiled WMMA kernel using v_wmma_f32_16x16x32_bf16
// (bf16 operands, fp32 accumulate): 128x64 block tile, 32x32 per wave
// (4 WMMA/K-step), double-buffered LDS fed by GLOBAL_LOAD_ASYNC_TO_LDS_B128
// (ASYNCcnt), one barrier per K-step. Fragment layouts per CDNA5 ISA 7.12.2.
// ---------------------------------------------------------------------------

typedef __attribute__((ext_vector_type(16))) __bf16 v16bf;
typedef __attribute__((ext_vector_type(8)))  __bf16 v8bf;
typedef __attribute__((ext_vector_type(8)))  float  v8f;

#define MB_  4
#define MT_  2048
#define MV_  1024
#define ML_  6
#define MD_  512
#define MH_  8
#define MDH_ 64
#define MFF_ 2048
#define MCB_ 256
#define MNC_ 4096
#define MQ_  2
#define MBT_ (MB_ * MT_)        // 8192
#define MBV_ (MB_ * MV_)        // 4096

__device__ __forceinline__ float gelu_tanh(float x) {
    const float k0 = 0.7978845608028654f;   // sqrt(2/pi)
    const float k1 = 0.044715f;
    return 0.5f * x * (1.0f + tanhf(k0 * (x + k1 * x * x * x)));
}

__device__ __forceinline__ v16bf cat8(v8bf lo, v8bf hi) {
    return __builtin_shufflevector(lo, hi, 0,1,2,3,4,5,6,7,8,9,10,11,12,13,14,15);
}

// CDNA5 async global->LDS copy (16 bytes / lane). VDST carries the 32-bit LDS
// byte address (low 32 bits of a generic LDS pointer == LDS offset, ISA 10.2),
// VADDR the 64-bit global address. Tracked by ASYNCcnt.
__device__ __forceinline__ void async_copy_b128(const __bf16* g, __bf16* l) {
    const unsigned      lds  = (unsigned)(uintptr_t)l;
    const unsigned long long ga = (unsigned long long)(uintptr_t)g;
    asm volatile("global_load_async_to_lds_b128 %0, %1, off"
                 :: "v"(lds), "v"(ga) : "memory");
}
__device__ __forceinline__ void async_wait0() {
    asm volatile("s_wait_asynccnt 0x0" ::: "memory");
}

// ---------------------------------------------------------------------------
// bf16 WMMA GEMM: C[M,N] = act(A[M,K] @ B + bias) + residual
//   - block = 256 threads (8 waves); block tile 128x64; wave tile 32x32
//   - BTRANS=0: B row-major [K,N]; BTRANS=1: B stored [N,K]
//   - B staged transposed in LDS so B fragments are contiguous reads
//   - requires M%128==0, N%64==0, K%32==0 (true for every call site here)
// ---------------------------------------------------------------------------
template <int BTRANS>
__global__ __launch_bounds__(256)
void gemm_bf16_wmma(const __bf16* __restrict__ A, int lda,
                    const __bf16* __restrict__ Bm, int ldb,
                    const float*  __restrict__ bias,
                    const float*  __restrict__ Rres, int ldr,
                    float*        __restrict__ Cf, int ldc,
                    __bf16*       __restrict__ Cb, int ldcb,
                    int M, int N, int K, int do_gelu)
{
    __shared__ __bf16 sA[2][128][40];   // [buf][m][k]; pad keeps 16B alignment
    __shared__ __bf16 sB[2][64][40];    // [buf][n][k] (transposed view of B)

    const int tid  = threadIdx.x;
    const int wave = tid >> 5;
    const int lane = tid & 31;
    const int half = lane >> 4;          // 0: lanes 0-15, 1: lanes 16-31
    const int r    = lane & 15;
    const int wm   = wave >> 1;          // 0..3 -> 32-row slabs
    const int wn   = wave & 1;           // 0..1 -> 32-col slabs
    const int m0   = blockIdx.y * 128;
    const int n0   = blockIdx.x * 64;

    v8f acc[2][2] = {};                  // [m-subtile][n-subtile]

    // A staging: 128x32 per buffer -> 16 bf16 per thread (2 x async b128)
    const int arow = tid >> 1;           // 0..127
    const int acol = (tid & 1) * 16;     // 0 or 16
    // B staging indices
    const int btn  = tid >> 2;           // BTRANS=1: 0..63 (n)
    const int btk  = (tid & 3) * 8;      //            k group
    const int bkk  = tid >> 3;           // BTRANS=0: 0..31 (k)
    const int bkn  = (tid & 7) * 8;      //            n group

    uint4 rb;                            // reg staging for BTRANS=0 transpose

    // Issue tile k0 into LDS buffer `buf`: async copies for A (and B when it
    // is contiguous along K); register staging for the B transpose path.
    auto issue_tile = [&](int k0, int buf) {
        const __bf16* asrc = A + (size_t)(m0 + arow) * lda + k0 + acol;
        async_copy_b128(asrc,     &sA[buf][arow][acol]);
        async_copy_b128(asrc + 8, &sA[buf][arow][acol + 8]);
        if (BTRANS) {
            async_copy_b128(Bm + (size_t)(n0 + btn) * ldb + k0 + btk,
                            &sB[buf][btn][btk]);
        } else {
            rb = *(const uint4*)(Bm + (size_t)(k0 + bkk) * ldb + n0 + bkn);
        }
    };
    // Complete tile into buffer `buf`: finish transpose scatter, drain
    // ASYNCcnt, publish to the workgroup.
    auto finish_tile = [&](int buf) {
        if (!BTRANS) {
            const __bf16* t = (const __bf16*)&rb;
            #pragma unroll
            for (int j = 0; j < 8; ++j) sB[buf][bkn + j][bkk] = t[j];
        }
        async_wait0();
        __syncthreads();
    };

    const int nsteps = K >> 5;
    issue_tile(0, 0);
    finish_tile(0);

    for (int s = 0; s < nsteps; ++s) {
        const int cur = s & 1;
        if (s + 1 < nsteps) {
            issue_tile((s + 1) << 5, 1 - cur);   // async into the other buffer
            if (s + 2 < nsteps)                  // prefetch the tile after
                __builtin_prefetch(A + (size_t)(m0 + arow) * lda +
                                   ((s + 2) << 5) + acol, 0, 3);
        }

        // ---- fragments (ISA 7.12.2 layouts) ----
        // A: row = slab + r ; K pairs {half*8.., 16+half*8..}
        const __bf16* ap0 = &sA[cur][wm * 32 + r][0];
        const __bf16* ap1 = &sA[cur][wm * 32 + 16 + r][0];
        const v16bf af0 = cat8(*(const v8bf*)(ap0 + half * 8),
                               *(const v8bf*)(ap0 + 16 + half * 8));
        const v16bf af1 = cat8(*(const v8bf*)(ap1 + half * 8),
                               *(const v8bf*)(ap1 + 16 + half * 8));
        // B: col = slab + r ; K pairs {half*8.., 16+half*8..}
        const __bf16* bp0 = &sB[cur][wn * 32 + r][0];
        const __bf16* bp1 = &sB[cur][wn * 32 + 16 + r][0];
        const v16bf bf0 = cat8(*(const v8bf*)(bp0 + half * 8), *(const v8bf*)(bp0 + 16 + half * 8));
        const v16bf bf1 = cat8(*(const v8bf*)(bp1 + half * 8), *(const v8bf*)(bp1 + 16 + half * 8));

        acc[0][0] = __builtin_amdgcn_wmma_f32_16x16x32_bf16(
            false, af0, false, bf0, (short)0, acc[0][0], false, false);
        acc[0][1] = __builtin_amdgcn_wmma_f32_16x16x32_bf16(
            false, af0, false, bf1, (short)0, acc[0][1], false, false);
        acc[1][0] = __builtin_amdgcn_wmma_f32_16x16x32_bf16(
            false, af1, false, bf0, (short)0, acc[1][0], false, false);
        acc[1][1] = __builtin_amdgcn_wmma_f32_16x16x32_bf16(
            false, af1, false, bf1, (short)0, acc[1][1], false, false);

        if (s + 1 < nsteps) {
            // safe: all reads of buf (1-cur) finished before the prior barrier
            finish_tile(1 - cur);
        }
    }

    // ---- epilogue: D layout -> vgpr i holds M = i + 8*half, N = r ----
    #pragma unroll
    for (int mi = 0; mi < 2; ++mi) {
        #pragma unroll
        for (int j = 0; j < 2; ++j) {
            const v8f accv = acc[mi][j];
            const int col  = n0 + wn * 32 + j * 16 + r;
            const float bv = bias ? bias[col] : 0.0f;
            #pragma unroll
            for (int i = 0; i < 8; ++i) {
                const int row = m0 + wm * 32 + mi * 16 + half * 8 + i;
                float v = accv[i] + bv;
                if (do_gelu) v = gelu_tanh(v);
                if (Rres)    v += Rres[(size_t)row * ldr + col];
                if (Cf) Cf[(size_t)row * ldc + col] = v;
                if (Cb) Cb[(size_t)row * ldcb + col] = (__bf16)v;
            }
        }
    }
}

// ---------------------------------------------------------------------------
// LayerNorm: one wave per row (wave32 shfl reductions), bf16 output
// ---------------------------------------------------------------------------
__global__ __launch_bounds__(256)
void layernorm_bf16(const float* __restrict__ x, const float* __restrict__ s,
                    const float* __restrict__ b, __bf16* __restrict__ out, int D)
{
    const int row  = blockIdx.x * 8 + (threadIdx.x >> 5);
    const int lane = threadIdx.x & 31;
    const float* xr = x + (size_t)row * D;
    float sum = 0.f;
    for (int i = lane; i < D; i += 32) sum += xr[i];
    #pragma unroll
    for (int o = 16; o; o >>= 1) sum += __shfl_xor(sum, o, 32);
    const float mean = sum / (float)D;
    float var = 0.f;
    for (int i = lane; i < D; i += 32) { const float d = xr[i] - mean; var += d * d; }
    #pragma unroll
    for (int o = 16; o; o >>= 1) var += __shfl_xor(var, o, 32);
    const float rstd = rsqrtf(var / (float)D + 1e-5f);
    __bf16* orow = out + (size_t)row * D;
    for (int i = lane; i < D; i += 32)
        orow[i] = (__bf16)((xr[i] - mean) * rstd * s[i] + b[i]);
}

// ---------------------------------------------------------------------------
// Masked softmax over score rows, bf16 probabilities out (wave per row)
// ---------------------------------------------------------------------------
__global__ __launch_bounds__(256)
void softmax_rows_bf16(const float* __restrict__ S,
                       const unsigned char* __restrict__ mask,
                       __bf16* __restrict__ P, int ncols, float scale)
{
    const int row  = blockIdx.x * 8 + (threadIdx.x >> 5);
    const int lane = threadIdx.x & 31;
    const float* sr = S + (size_t)row * ncols;
    float mx = -3.0e38f;
    for (int i = lane; i < ncols; i += 32) {
        const float v = sr[i] * scale + (mask[i] ? 0.0f : -1.0e9f);
        mx = fmaxf(mx, v);
    }
    #pragma unroll
    for (int o = 16; o; o >>= 1) mx = fmaxf(mx, __shfl_xor(mx, o, 32));
    float sum = 0.0f;
    for (int i = lane; i < ncols; i += 32) {
        const float v = sr[i] * scale + (mask[i] ? 0.0f : -1.0e9f);
        sum += __expf(v - mx);
    }
    #pragma unroll
    for (int o = 16; o; o >>= 1) sum += __shfl_xor(sum, o, 32);
    const float inv = 1.0f / sum;
    __bf16* pr = P + (size_t)row * ncols;
    for (int i = lane; i < ncols; i += 32) {
        const float v = sr[i] * scale + (mask[i] ? 0.0f : -1.0e9f);
        pr[i] = (__bf16)(__expf(v - mx) * inv);
    }
}

// ---------------------------------------------------------------------------
// Small utility kernels
// ---------------------------------------------------------------------------
__global__ void cvt_f32_bf16(const float* __restrict__ src,
                             __bf16* __restrict__ dst, size_t n)
{
    size_t i = (size_t)blockIdx.x * blockDim.x + threadIdx.x;
    const size_t st = (size_t)gridDim.x * blockDim.x;
    for (; i < n; i += st) dst[i] = (__bf16)src[i];
}

__global__ void zero_f32(float* __restrict__ p, size_t n)
{
    size_t i = (size_t)blockIdx.x * blockDim.x + threadIdx.x;
    const size_t st = (size_t)gridDim.x * blockDim.x;
    for (; i < n; i += st) p[i] = 0.0f;
}

__global__ void cnt_kernel(const int* __restrict__ faces,
                           const unsigned char* __restrict__ masks,
                           float* __restrict__ cnt)
{
    const int i = blockIdx.x * 256 + threadIdx.x;   // over B*T*3
    if (i >= MBT_ * 3) return;
    const int bt = i / 3;
    if (!masks[bt]) return;
    const int b = bt >> 11;                          // / T_
    atomicAdd(&cnt[b * MV_ + faces[i]], 1.0f);
}

__global__ void scatter_fe_kernel(const float* __restrict__ fe,
                                  const int* __restrict__ faces,
                                  const unsigned char* __restrict__ masks,
                                  float* __restrict__ sums)
{
    const int i = blockIdx.x * 256 + threadIdx.x;   // over B*T*3*CB
    const int c = i & (MCB_ - 1);
    const int corner_id = i >> 8;                    // b*T*3 + t*3 + corner
    const int bt = corner_id / 3;
    const int corner = corner_id - bt * 3;
    if (!masks[bt]) return;
    const int b = bt >> 11;
    const int flat = b * MV_ + faces[corner_id];
    atomicAdd(&sums[(size_t)flat * MCB_ + c],
              fe[(size_t)bt * (3 * MCB_) + corner * MCB_ + c]);
}

__global__ void vrtx_kernel(const float* __restrict__ sums,
                            const float* __restrict__ cnt,
                            float* __restrict__ res, __bf16* __restrict__ res_bf,
                            float* __restrict__ qo)
{
    const int i = blockIdx.x * 256 + threadIdx.x;   // over B*V*CB
    const int m = i >> 8;
    const float v = sums[i] / fmaxf(cnt[m], 1.0f);
    res[i] = v;
    res_bf[i] = (__bf16)v;
    qo[i] = 0.0f;
}

__global__ __launch_bounds__(256)
void rowsq_kernel(const float* __restrict__ cb, float* __restrict__ cb2, int cols)
{
    const int row  = blockIdx.x * 8 + (threadIdx.x >> 5);
    const int lane = threadIdx.x & 31;
    const float* cr = cb + (size_t)row * cols;
    float s = 0.f;
    for (int i = lane; i < cols; i += 32) { const float v = cr[i]; s += v * v; }
    #pragma unroll
    for (int o = 16; o; o >>= 1) s += __shfl_xor(s, o, 32);
    if (lane == 0) cb2[row] = s;
}

__global__ __launch_bounds__(256)
void argmin_kernel(const float* __restrict__ dot, const float* __restrict__ cb2,
                   int* __restrict__ idx, int N)
{
    const int row  = blockIdx.x * 8 + (threadIdx.x >> 5);
    const int lane = threadIdx.x & 31;
    const float* dr = dot + (size_t)row * N;
    float best = 3.0e38f;
    int   bi   = 0x7fffffff;
    for (int i = lane; i < N; i += 32) {
        const float d = cb2[i] - 2.0f * dr[i];   // + r^2 is row-constant
        if (d < best) { best = d; bi = i; }
    }
    #pragma unroll
    for (int o = 16; o; o >>= 1) {
        const float ob = __shfl_xor(best, o, 32);
        const int   oi = __shfl_xor(bi, o, 32);
        if (ob < best || (ob == best && oi < bi)) { best = ob; bi = oi; }
    }
    if (lane == 0) idx[row] = bi;
}

__global__ __launch_bounds__(256)
void vq_update_kernel(const float* __restrict__ codebook, const int* __restrict__ idx,
                      float* __restrict__ residual, __bf16* __restrict__ res_bf,
                      float* __restrict__ quant_out, float* __restrict__ commit_acc)
{
    __shared__ float red[256];
    const int gid = blockIdx.x * 256 + threadIdx.x;   // over B*V*CB
    const int m = gid >> 8;
    const int c = gid & (MCB_ - 1);
    const float q = codebook[(size_t)idx[m] * MCB_ + c];
    const float r = residual[gid];
    const float d = q - r;
    quant_out[gid] += q;                              // straight-through forward = quant
    const float nr = r - q;
    residual[gid] = nr;
    res_bf[gid] = (__bf16)nr;
    red[threadIdx.x] = d * d;
    __syncthreads();
    for (int s = 128; s; s >>= 1) {
        if (threadIdx.x < s) red[threadIdx.x] += red[threadIdx.x + s];
        __syncthreads();
    }
    if (threadIdx.x == 0) atomicAdd(commit_acc, red[0]);
}

__global__ void gather_kernel(const float* __restrict__ qo,
                              const int* __restrict__ faces,
                              const unsigned char* __restrict__ masks,
                              __bf16* __restrict__ gq)
{
    const int i = blockIdx.x * 256 + threadIdx.x;     // over B*T*3*CB
    const int c = i & (MCB_ - 1);
    const int corner_id = i >> 8;
    const int bt = corner_id / 3;
    const int corner = corner_id - bt * 3;
    float v = 0.0f;
    if (masks[bt]) {
        const int b = bt >> 11;
        const int flat = b * MV_ + faces[corner_id];
        v = qo[(size_t)flat * MCB_ + c];
    }
    gq[(size_t)bt * (3 * MCB_) + corner * MCB_ + c] = (__bf16)v;
}

__global__ void codes_out_kernel(const int* __restrict__ faces,
                                 const unsigned char* __restrict__ masks,
                                 const int* __restrict__ idx0,
                                 const int* __restrict__ idx1,
                                 float* __restrict__ out)
{
    const int i = blockIdx.x * 256 + threadIdx.x;     // over B*T*3*Q
    if (i >= MBT_ * 3 * MQ_) return;
    const int qi = i & 1;
    const int corner_id = i >> 1;
    const int bt = corner_id / 3;
    float v = 0.0f;
    if (masks[bt]) {
        const int b = bt >> 11;
        const int flat = b * MV_ + faces[corner_id];
        v = (float)(qi ? idx1[flat] : idx0[flat]);
    }
    out[i] = v;
}

__global__ void commit_kernel(const float* __restrict__ acc, float* __restrict__ out)
{
    if (threadIdx.x == 0 && blockIdx.x == 0)
        out[0] = acc[0] / (float)((size_t)MBV_ * MCB_);
}

// ---------------------------------------------------------------------------
// Host orchestration
// ---------------------------------------------------------------------------
static inline size_t AL(size_t x) { return (x + 255) & ~(size_t)255; }

static inline void gemm(hipStream_t st, const __bf16* A, int lda,
                        const __bf16* Bm, int ldb, int btrans,
                        const float* bias, const float* res, int ldr,
                        float* Cf, int ldc, __bf16* Cb, int ldcb,
                        int M, int N, int K, int do_gelu)
{
    dim3 g(N / 64, M / 128);
    if (btrans)
        gemm_bf16_wmma<1><<<g, 256, 0, st>>>(A, lda, Bm, ldb, bias, res, ldr,
                                             Cf, ldc, Cb, ldcb, M, N, K, do_gelu);
    else
        gemm_bf16_wmma<0><<<g, 256, 0, st>>>(A, lda, Bm, ldb, bias, res, ldr,
                                             Cf, ldc, Cb, ldcb, M, N, K, do_gelu);
}

extern "C" void kernel_launch(void* const* d_in, const int* in_sizes, int n_in,
                              void* d_out, int out_size, void* d_ws, size_t ws_size,
                              hipStream_t stream)
{
    (void)in_sizes; (void)n_in; (void)out_size; (void)ws_size;

    // inputs (setup_inputs order)
    const int*           faces  = (const int*)d_in[0];
    const float*         fembed = (const float*)d_in[1];
    const unsigned char* masks  = (const unsigned char*)d_in[2];
    const float* ln1_s = (const float*)d_in[3];
    const float* ln1_b = (const float*)d_in[4];
    const float* wq    = (const float*)d_in[5];
    const float* wk    = (const float*)d_in[6];
    const float* wv    = (const float*)d_in[7];
    const float* wo    = (const float*)d_in[8];
    const float* ln2_s = (const float*)d_in[9];
    const float* ln2_b = (const float*)d_in[10];
    const float* w1    = (const float*)d_in[11];
    const float* b1    = (const float*)d_in[12];
    const float* w2    = (const float*)d_in[13];
    const float* b2    = (const float*)d_in[14];
    const float* lnf_s = (const float*)d_in[15];
    const float* lnf_b = (const float*)d_in[16];
    const float* pv_w  = (const float*)d_in[17];
    const float* pv_b  = (const float*)d_in[18];
    const float* pl_w  = (const float*)d_in[19];
    const float* pl_b  = (const float*)d_in[20];
    const float* cbk   = (const float*)d_in[21];

    // workspace carve-out
    char* ws = (char*)d_ws;
    size_t cur = 0;
    #define CARVE(ptr_t, name, bytes) ptr_t name = (ptr_t)(ws + cur); cur = AL(cur + (bytes))
    CARVE(float*,  X,    (size_t)MBT_ * MD_ * 4);           // residual stream fp32
    CARVE(__bf16*, HB,   (size_t)MBT_ * MD_ * 2);           // LN output bf16
    CARVE(__bf16*, QB,   (size_t)MBT_ * MD_ * 2);
    CARVE(__bf16*, KB,   (size_t)MBT_ * MD_ * 2);
    CARVE(__bf16*, VB,   (size_t)MBT_ * MD_ * 2);
    CARVE(float*,  SC,   (size_t)MT_ * MT_ * 4);            // scores (reused per head)
    CARVE(__bf16*, PB,   (size_t)MT_ * MT_ * 2);            // probs bf16
    CARVE(__bf16*, AB,   (size_t)MBT_ * MD_ * 2);           // attn out bf16
    CARVE(__bf16*, FFB,  (size_t)MBT_ * MFF_ * 2);          // gelu(h@w1+b1) bf16
    CARVE(float*,  FE,   (size_t)MBT_ * 3 * MCB_ * 4);      // pv projection fp32
    CARVE(__bf16*, WQW,  (size_t)ML_ * MD_ * MD_ * 2);
    CARVE(__bf16*, WKW,  (size_t)ML_ * MD_ * MD_ * 2);
    CARVE(__bf16*, WVW,  (size_t)ML_ * MD_ * MD_ * 2);
    CARVE(__bf16*, WOW,  (size_t)ML_ * MD_ * MD_ * 2);
    CARVE(__bf16*, WW1,  (size_t)ML_ * MD_ * MFF_ * 2);
    CARVE(__bf16*, WW2,  (size_t)ML_ * MFF_ * MD_ * 2);
    CARVE(__bf16*, PVW,  (size_t)MD_ * 3 * MCB_ * 2);
    CARVE(__bf16*, PLW,  (size_t)3 * MCB_ * MD_ * 2);
    CARVE(__bf16*, CBB,  (size_t)MNC_ * MCB_ * 2);          // codebook bf16
    CARVE(float*,  SUMS, (size_t)MBV_ * MCB_ * 4);
    CARVE(float*,  CNT,  (size_t)MBV_ * 4);
    CARVE(float*,  RES,  (size_t)MBV_ * MCB_ * 4);
    CARVE(__bf16*, RESB, (size_t)MBV_ * MCB_ * 2);
    CARVE(float*,  QO,   (size_t)MBV_ * MCB_ * 4);
    CARVE(float*,  DOT,  (size_t)MBV_ * MNC_ * 4);
    CARVE(float*,  CB2,  (size_t)MNC_ * 4);
    CARVE(int*,    IDX0, (size_t)MBV_ * 4);
    CARVE(int*,    IDX1, (size_t)MBV_ * 4);
    CARVE(float*,  CACC, 256);
    CARVE(__bf16*, GQB,  (size_t)MBT_ * 3 * MCB_ * 2);
    #undef CARVE

    float* out_face   = (float*)d_out;
    float* out_codes  = out_face + (size_t)MBT_ * MD_;
    float* out_commit = out_codes + (size_t)MBT_ * 3 * MQ_;

    // --- one-time conversions / zeroing (re-done every call: deterministic) ---
    cvt_f32_bf16<<<2048, 256, 0, stream>>>(wq,   WQW, (size_t)ML_ * MD_ * MD_);
    cvt_f32_bf16<<<2048, 256, 0, stream>>>(wk,   WKW, (size_t)ML_ * MD_ * MD_);
    cvt_f32_bf16<<<2048, 256, 0, stream>>>(wv,   WVW, (size_t)ML_ * MD_ * MD_);
    cvt_f32_bf16<<<2048, 256, 0, stream>>>(wo,   WOW, (size_t)ML_ * MD_ * MD_);
    cvt_f32_bf16<<<2048, 256, 0, stream>>>(w1,   WW1, (size_t)ML_ * MD_ * MFF_);
    cvt_f32_bf16<<<2048, 256, 0, stream>>>(w2,   WW2, (size_t)ML_ * MFF_ * MD_);
    cvt_f32_bf16<<<2048, 256, 0, stream>>>(pv_w, PVW, (size_t)MD_ * 3 * MCB_);
    cvt_f32_bf16<<<2048, 256, 0, stream>>>(pl_w, PLW, (size_t)3 * MCB_ * MD_);
    cvt_f32_bf16<<<2048, 256, 0, stream>>>(cbk,  CBB, (size_t)MNC_ * MCB_);
    zero_f32<<<1024, 256, 0, stream>>>(SUMS, (size_t)MBV_ * MCB_);
    zero_f32<<<16,   256, 0, stream>>>(CNT,  (size_t)MBV_);
    zero_f32<<<1,    256, 0, stream>>>(CACC, 64);

    const float scale = 0.125f;   // 1/sqrt(DH)

    // --- transformer encoder ---
    for (int l = 0; l < ML_; ++l) {
        const float* xin = (l == 0) ? fembed : X;
        layernorm_bf16<<<MBT_ / 8, 256, 0, stream>>>(xin, ln1_s + l * MD_,
                                                     ln1_b + l * MD_, HB, MD_);
        gemm(stream, HB, MD_, WQW + (size_t)l * MD_ * MD_, MD_, 0,
             nullptr, nullptr, 0, nullptr, 0, QB, MD_, MBT_, MD_, MD_, 0);
        gemm(stream, HB, MD_, WKW + (size_t)l * MD_ * MD_, MD_, 0,
             nullptr, nullptr, 0, nullptr, 0, KB, MD_, MBT_, MD_, MD_, 0);
        gemm(stream, HB, MD_, WVW + (size_t)l * MD_ * MD_, MD_, 0,
             nullptr, nullptr, 0, nullptr, 0, VB, MD_, MBT_, MD_, MD_, 0);

        for (int b = 0; b < MB_; ++b) {
            for (int h = 0; h < MH_; ++h) {
                const size_t hd = (size_t)b * MT_ * MD_ + (size_t)h * MDH_;
                // S = Q @ K^T  (K rows are "N", so B-transposed)
                gemm(stream, QB + hd, MD_, KB + hd, MD_, 1,
                     nullptr, nullptr, 0, SC, MT_, nullptr, 0,
                     MT_, MT_, MDH_, 0);
                softmax_rows_bf16<<<MT_ / 8, 256, 0, stream>>>(
                    SC, masks + (size_t)b * MT_, PB, MT_, scale);
                // O_head = P @ V  -> strided bf16 into AB
                gemm(stream, PB, MT_, VB + hd, MD_, 0,
                     nullptr, nullptr, 0, nullptr, 0, AB + hd, MD_,
                     MT_, MDH_, MT_, 0);
            }
        }
        // x = x + attn @ Wo
        gemm(stream, AB, MD_, WOW + (size_t)l * MD_ * MD_, MD_, 0,
             nullptr, xin, MD_, X, MD_, nullptr, 0, MBT_, MD_, MD_, 0);

        layernorm_bf16<<<MBT_ / 8, 256, 0, stream>>>(X, ln2_s + l * MD_,
                                                     ln2_b + l * MD_, HB, MD_);
        // ff = gelu(h @ W1 + b1)  (bf16 out for next GEMM)
        gemm(stream, HB, MD_, WW1 + (size_t)l * MD_ * MFF_, MFF_, 0,
             b1 + (size_t)l * MFF_, nullptr, 0, nullptr, 0, FFB, MFF_,
             MBT_, MFF_, MD_, 1);
        // x = x + ff @ W2 + b2
        gemm(stream, FFB, MFF_, WW2 + (size_t)l * MFF_ * MD_, MD_, 0,
             b2 + (size_t)l * MD_, X, MD_, X, MD_, nullptr, 0,
             MBT_, MD_, MFF_, 0);
    }

    // --- final LN + vertex projection ---
    layernorm_bf16<<<MBT_ / 8, 256, 0, stream>>>(X, lnf_s, lnf_b, HB, MD_);
    gemm(stream, HB, MD_, PVW, 3 * MCB_, 0,
         pv_b, nullptr, 0, FE, 3 * MCB_, nullptr, 0, MBT_, 3 * MCB_, MD_, 0);

    // --- scatter-mean into shared vertices ---
    cnt_kernel<<<(MBT_ * 3 + 255) / 256, 256, 0, stream>>>(faces, masks, CNT);
    scatter_fe_kernel<<<(MBT_ * 3 * MCB_) / 256, 256, 0, stream>>>(FE, faces, masks, SUMS);
    vrtx_kernel<<<(MBV_ * MCB_) / 256, 256, 0, stream>>>(SUMS, CNT, RES, RESB, QO);

    // --- residual VQ (Q = 2, shared codebook) ---
    rowsq_kernel<<<MNC_ / 8, 256, 0, stream>>>(cbk, CB2, MCB_);
    int* idxs[MQ_] = { IDX0, IDX1 };
    for (int q = 0; q < MQ_; ++q) {
        // dot = residual @ codebook^T  (codebook stored [NC, CB] -> B-transposed)
        gemm(stream, RESB, MCB_, CBB, MCB_, 1,
             nullptr, nullptr, 0, DOT, MNC_, nullptr, 0, MBV_, MNC_, MCB_, 0);
        argmin_kernel<<<MBV_ / 8, 256, 0, stream>>>(DOT, CB2, idxs[q], MNC_);
        vq_update_kernel<<<(MBV_ * MCB_) / 256, 256, 0, stream>>>(
            cbk, idxs[q], RES, RESB, QO, CACC);
    }

    // --- gather quantized corners, project back, emit codes + commit ---
    gather_kernel<<<(MBT_ * 3 * MCB_) / 256, 256, 0, stream>>>(QO, faces, masks, GQB);
    gemm(stream, GQB, 3 * MCB_, PLW, MD_, 0,
         pl_b, nullptr, 0, out_face, MD_, nullptr, 0, MBT_, MD_, 3 * MCB_, 0);
    codes_out_kernel<<<(MBT_ * 3 * MQ_ + 255) / 256, 256, 0, stream>>>(
        faces, masks, IDX0, IDX1, out_codes);
    commit_kernel<<<1, 32, 0, stream>>>(CACC, out_commit);
}
